// MinimalConvLSTM_29515015258259
// MI455X (gfx1250) — matrix operand order repaired and checked by reading, not compiled, not run on hardware
//
#include <hip/hip_runtime.h>

typedef _Float16 half_t;
typedef __attribute__((ext_vector_type(16))) _Float16 v16h;
typedef __attribute__((ext_vector_type(8)))  _Float16 v8h;
typedef __attribute__((ext_vector_type(8)))  float    v8f;

#define B_    8
#define T_    12
#define H_    128
#define W_    128
#define F_    64
#define NG_   256          // 4*F
#define PRED_ 6
#define KCH_  19           // 18 recurrent K-chunks (K=576) + 1 padded input chunk (K=9 of 32)
#define TILE_X 32
#define HALO_X 34

#if __has_builtin(__builtin_amdgcn_sched_group_barrier)
#define SCHED_GROUP(mask, n, id) __builtin_amdgcn_sched_group_barrier((mask), (n), (id))
#else
#define SCHED_GROUP(mask, n, id)
#endif
// sched_group masks: 0x8 = MFMA/WMMA, 0x20 = VMEM read, 0x100 = DS read

__device__ __forceinline__ float sigmoidf_(float v) {
  return 1.0f / (1.0f + __expf(-v));
}
__device__ __forceinline__ float tanhf_(float v) {
  return 2.0f * sigmoidf_(2.0f * v) - 1.0f;   // 1 hardware exp per tanh
}

// ---------------------------------------------------------------------------
// Prep: pack rec_kernel (3,3,64,256) + kernel (3,3,1,256) into fp16 WMMA
// B-fragment order: wpack[c][ntile][lane][16 halves]
//   lane<16 : N=lane,    k_local = e<8 ? e : 16+(e-8)
//   lane>=16: N=lane-16, k_local = e<8 ? 8+e : 24+(e-8)
//   c<18 : (ky,kx)=c>>1, ci=(c&1)*32+k_local ; c==18: input conv, k_local<9 valid
// ---------------------------------------------------------------------------
__global__ __launch_bounds__(256) void pack_weights_kernel(
    const float* __restrict__ kern, const float* __restrict__ rec,
    half_t* __restrict__ wpack) {
  int idx = blockIdx.x * 256 + threadIdx.x;
  const int total = KCH_ * 16 * 32 * 16;
  if (idx >= total) return;
  int e    = idx & 15;
  int lane = (idx >> 4) & 31;
  int nt   = (idx >> 9) & 15;
  int c    = idx >> 13;
  int n    = lane & 15;
  int kl   = (e < 8) ? ((lane >> 4) << 3) + e : 16 + ((lane >> 4) << 3) + (e - 8);   // k_local 0..31
  int co   = nt * 16 + n;
  float v = 0.0f;
  if (c < 18) {
    int w9 = c >> 1;                       // (ky*3+kx)
    int ci = ((c & 1) << 5) + kl;
    v = rec[((size_t)w9 * F_ + ci) * NG_ + co];
  } else if (kl < 9) {
    v = kern[(size_t)kl * NG_ + co];
  }
  wpack[idx] = (half_t)v;
}

__global__ __launch_bounds__(256) void cvt_x_kernel(
    const float* __restrict__ x, half_t* __restrict__ x16, int n) {
  int i = blockIdx.x * 256 + threadIdx.x;
  if (i < n) x16[i] = (half_t)x[i];
}

__global__ __launch_bounds__(256) void zero_state_kernel(
    half_t* __restrict__ h, float* __restrict__ c, int n) {
  int i = blockIdx.x * 256 + threadIdx.x;
  if (i < n) { h[i] = (half_t)0.0f; c[i] = 0.0f; }
}

// ---------------------------------------------------------------------------
// One ConvLSTM time step: z = conv3x3(x_t)+conv3x3(h)+bias as a WMMA GEMM
// (M=32 positions per block, N=256 gates, K=576+9), then gate math in regs.
// Block: 256 threads = 8 waves. Wave (wm, col): M-tile wm (16 x-positions),
// gate column-group col; its 4 accumulators are one full {i,f,g,o} gate
// quadruple. B fragments: triple-buffered, distance-2 prefetch; per-chunk
// schedule pinned with WMMAs *before* the prefetch loads:
//   [4 WMMA (c)] [8 VMEM (c+2)] [2 DS (c+1)]
// ---------------------------------------------------------------------------
__global__ __launch_bounds__(256) void lstm_gate_kernel(
    const half_t* __restrict__ x16,   // (B,T,H,W) fp16
    int t,
    const half_t* __restrict__ hin,   // (B,H,W,F) fp16
    const float*  __restrict__ cin,   // (B,H,W,F) f32
    const half_t* __restrict__ wpack, // [19][16][32][16] fp16
    const float*  __restrict__ bias,  // (256)
    half_t* __restrict__ hout,
    float*  __restrict__ cout_) {
  __shared__ __align__(16) half_t sh[3][HALO_X][F_];  // h halo, 13056 B
  __shared__ half_t sx[3][HALO_X];                    // x_t halo

  const int tid = threadIdx.x;
  const int x0  = blockIdx.x * TILE_X;
  const int y   = blockIdx.y;
  const int b   = blockIdx.z;

  // ---- stage halos into LDS (16B vector loads for h) ----
  {
    const int nvec = 3 * HALO_X * (F_ / 8);  // 816
    for (int v = tid; v < nvec; v += 256) {
      int c8 = (v % (F_ / 8)) * 8;
      int xh = (v / (F_ / 8)) % HALO_X;
      int r  = v / ((F_ / 8) * HALO_X);
      int gy = y + r - 1;
      int gx = x0 + xh - 1;
      v8h val;
      if (gy >= 0 && gy < H_ && gx >= 0 && gx < W_) {
        val = *(const v8h*)&hin[(((size_t)b * H_ + gy) * W_ + gx) * F_ + c8];
      } else {
        #pragma unroll
        for (int e = 0; e < 8; ++e) val[e] = (half_t)0.0f;
      }
      *(v8h*)&sh[r][xh][c8] = val;
    }
    for (int v = tid; v < 3 * HALO_X; v += 256) {
      int xh = v % HALO_X;
      int r  = v / HALO_X;
      int gy = y + r - 1;
      int gx = x0 + xh - 1;
      half_t val = (half_t)0.0f;
      if (gy >= 0 && gy < H_ && gx >= 0 && gx < W_)
        val = x16[(((size_t)b * T_ + t) * H_ + gy) * W_ + gx];
      sx[r][xh] = val;
    }
  }
  __syncthreads();

  const int wave = tid >> 5;
  const int lane = tid & 31;
  const int wm   = wave >> 2;   // M tile 0..1
  const int col  = wave & 3;    // gate column-group 0..3
  const int mlo  = lane & 15;   // A-fragment row M
  const int hi   = lane >> 4;

  v8f acc[4];
  #pragma unroll
  for (int j = 0; j < 4; ++j)
    #pragma unroll
    for (int r = 0; r < 8; ++r) acc[j][r] = 0.0f;

  // B fragment loader: 4 gate n-tiles of chunk c -> bf[0..3] (8 coalesced b128s)
  auto loadB = [&](int c, v16h* bf) {
    const half_t* bbase = wpack + (((size_t)c * 16) * 32 + lane) * 16;
    #pragma unroll
    for (int j = 0; j < 4; ++j) {
      const int nt = j * 4 + col;            // gate j, column group col
      bf[j] = *(const v16h*)(bbase + (size_t)nt * 32 * 16);
    }
  };
  // A fragment loader (chunk c)
  auto loadA = [&](int c) -> v16h {
    v16h af;
    if (c < 18) {
      const int w9  = c >> 1;
      const int ky  = w9 / 3, kx = w9 % 3;
      const int cib = ((c & 1) << 5) + hi * 8;   // ci base for this lane group
      const half_t* ap = &sh[ky][wm * 16 + mlo + kx][cib];
      v8h alo = *(const v8h*)ap;          // k_local = hi*8 + 0..7
      v8h ahi = *(const v8h*)(ap + 16);   // k_local = hi*8 + 16..23
      #pragma unroll
      for (int e = 0; e < 8; ++e) { af[e] = alo[e]; af[8 + e] = ahi[e]; }
    } else {
      // input conv chunk: K = 9 valid of 32
      #pragma unroll
      for (int e = 0; e < 16; ++e) af[e] = (half_t)0.0f;
      #pragma unroll
      for (int e = 0; e < 8; ++e) {
        int kp = hi * 8 + e;
        if (kp < 9) af[e] = sx[kp / 3][wm * 16 + mlo + kp % 3];
      }
    }
    return af;
  };

  // ---- GEMM over 19 K-chunks; B triple-buffered (distance-2 prefetch),
  //      A prefetched one chunk ahead. WMMAs come FIRST in both program
  //      order and the sched-group template so the prefetch cannot be
  //      rotated back onto the consuming chunk. ----
  v16h bbuf[3][4];
  loadB(0, bbuf[0]);
  loadB(1, bbuf[1]);
  v16h anext = loadA(0);
  #pragma unroll
  for (int c = 0; c < KCH_; ++c) {
    v16h acur = anext;
    v16h* bc = bbuf[c % 3];
    #pragma unroll
    for (int j = 0; j < 4; ++j)
      acc[j] = __builtin_amdgcn_wmma_f32_16x16x32_f16(
          false, acur, false, bc[j], (short)0, acc[j], false, false);
    if (c + 2 < KCH_) loadB(c + 2, bbuf[(c + 2) % 3]);
    if (c + 1 < KCH_) anext = loadA(c + 1);
    // Pin per-chunk order: WMMAs (c) -> B prefetch (c+2) -> A prefetch (c+1)
    SCHED_GROUP(0x008, 4, 0);   // 4 WMMA
    SCHED_GROUP(0x020, 8, 0);   // 8 VMEM reads
    SCHED_GROUP(0x100, 2, 0);   // 2 DS reads
  }

  // ---- LSTM cell update entirely in registers ----
  const int n   = lane & 15;
  const int fch = col * 16 + n;
  const float bi = bias[0 * F_ + fch];
  const float bf = bias[1 * F_ + fch];
  const float bg = bias[2 * F_ + fch];
  const float bo = bias[3 * F_ + fch];
  #pragma unroll
  for (int r = 0; r < 8; ++r) {
    const int px = x0 + wm * 16 + r + 8 * hi;   // D layout: M = r + 8*(lane>=16)
    const size_t idx = (((size_t)b * H_ + y) * W_ + px) * F_ + fch;
    float zi = acc[0][r] + bi;
    float zf = acc[1][r] + bf;
    float zg = acc[2][r] + bg;
    float zo = acc[3][r] + bo;
    float cold = cin[idx];
    float cn = sigmoidf_(zf) * cold + sigmoidf_(zi) * tanhf_(zg);
    float hn = sigmoidf_(zo) * tanhf_(cn);
    cout_[idx] = cn;
    hout[idx]  = (half_t)hn;
  }
}

// ---------------------------------------------------------------------------
// Decoder: pred = sigmoid(conv3x3(cur, w_out) + b_out)  (tiny: N=1)
// ---------------------------------------------------------------------------
__global__ __launch_bounds__(256) void pred_kernel(
    const half_t* __restrict__ cur,   // (B,H,W,F)
    const float* __restrict__ w_out,  // (3,3,F,1)
    const float* __restrict__ b_out,
    float* __restrict__ out,          // (B,PRED,H,W)
    int s) {
  __shared__ float sw[9 * F_];
  const int tid = threadIdx.x;
  for (int i = tid; i < 9 * F_; i += 256) sw[i] = w_out[i];
  __syncthreads();
  int p = blockIdx.x * 256 + tid;
  int x = p % W_;
  int y = (p / W_) % H_;
  int b = p / (W_ * H_);
  float accv = b_out[0];
  for (int ky = 0; ky < 3; ++ky) {
    int gy = y + ky - 1;
    if (gy < 0 || gy >= H_) continue;
    for (int kx = 0; kx < 3; ++kx) {
      int gx = x + kx - 1;
      if (gx < 0 || gx >= W_) continue;
      const half_t* hp = &cur[(((size_t)b * H_ + gy) * W_ + gx) * F_];
      const float* wp = &sw[(ky * 3 + kx) * F_];
      #pragma unroll 8
      for (int ci = 0; ci < F_; ++ci) accv += (float)hp[ci] * wp[ci];
    }
  }
  out[(((size_t)b * PRED_ + s) * H_ + y) * W_ + x] = sigmoidf_(accv);
}

// cur = relu(conv1x1(pred, w_proj) + b_proj)
__global__ __launch_bounds__(256) void proj_kernel(
    const float* __restrict__ out, int s,
    const float* __restrict__ w_proj, const float* __restrict__ b_proj,
    half_t* __restrict__ cur) {
  int i = blockIdx.x * 256 + threadIdx.x;   // over B*H*W*F
  int ch = i & (F_ - 1);
  int p  = i >> 6;
  int x = p % W_;
  int y = (p / W_) % H_;
  int b = p / (W_ * H_);
  float pv = out[(((size_t)b * PRED_ + s) * H_ + y) * W_ + x];
  float v  = pv * w_proj[ch] + b_proj[ch];
  cur[(size_t)p * F_ + ch] = (half_t)fmaxf(v, 0.0f);
}

// ---------------------------------------------------------------------------
extern "C" void kernel_launch(void* const* d_in, const int* in_sizes, int n_in,
                              void* d_out, int out_size, void* d_ws, size_t ws_size,
                              hipStream_t stream) {
  const float* x      = (const float*)d_in[0];
  const float* kern   = (const float*)d_in[1];
  const float* rec    = (const float*)d_in[2];
  const float* bias   = (const float*)d_in[3];
  const float* w_out  = (const float*)d_in[4];
  const float* b_out  = (const float*)d_in[5];
  const float* w_proj = (const float*)d_in[6];
  const float* b_proj = (const float*)d_in[7];
  float* out = (float*)d_out;

  const size_t NX  = (size_t)B_ * T_ * H_ * W_;          // 1.57M
  const size_t NHC = (size_t)B_ * H_ * W_ * F_;          // 8.39M
  const size_t NPK = (size_t)KCH_ * 16 * 32 * 16;        // 155648

  char* p = (char*)d_ws;
  auto carve = [&](size_t bytes) {
    char* r = p;
    p += (bytes + 255) & ~(size_t)255;
    return r;
  };
  half_t* x16   = (half_t*)carve(NX * sizeof(half_t));
  half_t* wpack = (half_t*)carve(NPK * sizeof(half_t));
  half_t* hA    = (half_t*)carve(NHC * sizeof(half_t));
  half_t* hB    = (half_t*)carve(NHC * sizeof(half_t));
  float*  cA    = (float*)carve(NHC * sizeof(float));
  float*  cB    = (float*)carve(NHC * sizeof(float));
  half_t* cur   = (half_t*)carve(NHC * sizeof(half_t));

  pack_weights_kernel<<<(int)((NPK + 255) / 256), 256, 0, stream>>>(kern, rec, wpack);
  cvt_x_kernel<<<(int)((NX + 255) / 256), 256, 0, stream>>>(x, x16, (int)NX);
  zero_state_kernel<<<(int)((NHC + 255) / 256), 256, 0, stream>>>(hA, cA, (int)NHC);

  half_t* hin = hA; half_t* hout = hB;
  float*  cinp = cA; float* coutp = cB;
  dim3 g(W_ / TILE_X, H_, B_);
  for (int t = 0; t < T_; ++t) {
    lstm_gate_kernel<<<g, 256, 0, stream>>>(x16, t, hin, cinp, wpack, bias, hout, coutp);
    half_t* th = hin; hin = hout; hout = th;
    float*  tc = cinp; cinp = coutp; coutp = tc;
  }

  const half_t* curp = hin;  // final hidden state
  for (int s = 0; s < PRED_; ++s) {
    pred_kernel<<<(B_ * H_ * W_ + 255) / 256, 256, 0, stream>>>(curp, w_out, b_out, out, s);
    if (s + 1 < PRED_) {
      proj_kernel<<<(int)((NHC + 255) / 256), 256, 0, stream>>>(out, s, w_proj, b_proj, cur);
      curp = cur;
    }
  }
  (void)in_sizes; (void)n_in; (void)out_size; (void)ws_size;
}
